// DeformableTransformerDecoderLayer_61272003444831
// MI455X (gfx1250) — matrix operand order repaired and checked by reading, not compiled, not run on hardware
//
#include <hip/hip_runtime.h>
#include <hip/hip_bf16.h>

#define DEV __device__ __forceinline__

typedef __attribute__((ext_vector_type(16))) __bf16 v16bf;
typedef __attribute__((ext_vector_type(8)))  float  v8f;
typedef __attribute__((ext_vector_type(4)))  unsigned int u32x4;
typedef __attribute__((ext_vector_type(4)))  int i32x4;
typedef __attribute__((ext_vector_type(8)))  int i32x8;

constexpr int D  = 256, NH = 8, NL = 4, NPT = 4, DFF = 1024, BS = 8, NQ = 900;
constexpr int LS = 13294;
constexpr int NQP = 928;                    // key length padded to multiple of 32
constexpr size_t OUT_LOC = (size_t)BS * NQ * D;                 // 1,843,200
constexpr size_t OUT_AW  = OUT_LOC + (size_t)BS * NQ * NH * NL * NPT * 2; // +1,843,200

struct alignas(16) U4 { unsigned x, y, z, w; };
union FragU { U4 q[2]; v16bf v; };

DEV v8f vzero() {
  v8f v;
#pragma unroll
  for (int i = 0; i < 8; ++i) v[i] = 0.f;
  return v;
}

DEV float wsum(float v) {
#pragma unroll
  for (int o = 16; o >= 1; o >>= 1) v += __shfl_xor(v, o, 32);
  return v;
}

DEV float wmax(float v) {
#pragma unroll
  for (int o = 16; o >= 1; o >>= 1) v = fmaxf(v, __shfl_xor(v, o, 32));
  return v;
}

// ---------------- CDNA5 Tensor Data Mover: 2D bf16 tile -> LDS (D# per ISA 08 §8)
DEV void tdm_load_2d_bf16(unsigned lds_addr, const __bf16* gptr,
                          unsigned remW, unsigned remH,
                          unsigned tileW, unsigned tileH, unsigned strideElems) {
  u32x4 g0; i32x8 g1; i32x4 gz; i32x8 gz8;
#pragma unroll
  for (int i = 0; i < 4; ++i) gz[i] = 0;
#pragma unroll
  for (int i = 0; i < 8; ++i) gz8[i] = 0;
  unsigned long long ga = (unsigned long long)(size_t)gptr;
  // group0: count=1 | lds_addr | global_addr[56:0] | type=2 ("image")
  g0[0] = 1u;
  g0[1] = lds_addr;
  g0[2] = (unsigned)ga;
  g0[3] = (unsigned)((ga >> 32) & 0x01ffffffu) | (2u << 30);
  // group1: data_size=1 (2B); tensor_dim0/1 = remaining extents; tile dims; dim0 stride
  unsigned long long st = strideElems;
  g1[0] = (int)(1u << 16);
  g1[1] = (int)((remW & 0xffffu) << 16);
  g1[2] = (int)(((remW >> 16) & 0xffffu) | ((remH & 0xffffu) << 16));
  g1[3] = (int)(((remH >> 16) & 0xffffu) | ((tileW & 0xffffu) << 16));
  g1[4] = (int)(tileH & 0xffffu);                 // tile_dim1; tile_dim2 = 0
  g1[5] = (int)(unsigned)(st & 0xffffffffu);      // tensor_dim0_stride lo
  g1[6] = (int)((st >> 32) & 0xffffu);            // tensor_dim0_stride hi
  g1[7] = 0;
#if defined(__clang_major__) && __clang_major__ >= 23
  __builtin_amdgcn_tensor_load_to_lds(g0, g1, gz, gz, gz8, 0);
#else
  (void)gz8;
  __builtin_amdgcn_tensor_load_to_lds(g0, g1, gz, gz, 0);
#endif
}

// ---------------------------------------------------------------- converts
__global__ void k_f32_to_bf16(const float* __restrict__ in, __bf16* __restrict__ out, size_t n) {
  size_t i = (size_t)blockIdx.x * blockDim.x + threadIdx.x;
  if (i < n) out[i] = (__bf16)in[i];
}

// q = (tgt+qpos) transposed to [NQ,BS,D]; tgtT = tgt transposed. Both bf16.
__global__ void k_prep_q(const float* __restrict__ tgt, const float* __restrict__ qpos,
                         __bf16* __restrict__ q_bf, __bf16* __restrict__ tgtT_bf) {
  size_t i = (size_t)blockIdx.x * blockDim.x + threadIdx.x;
  if (i >= (size_t)BS * NQ * D) return;
  int d = (int)(i % D); size_t r = i / D; int n = (int)(r % NQ); int b = (int)(r / NQ);
  float tv = tgt[i];
  size_t o = ((size_t)n * BS + b) * D + d;
  q_bf[o]    = (__bf16)(tv + qpos[i]);
  tgtT_bf[o] = (__bf16)tv;
}

// ---------------------------------------------------------------- WMMA GEMM
// C[M,N] = A[M,K](bf16) * B[N,K]^T(bf16) + bias. Optional relu, f32/bf16 out,
// and "attention scatter" to [b, NH, seq, hd] (f32 and/or bf16).
// TDM=true: double-buffered Tensor-Data-Mover DMA of A/B tiles into LDS,
// overlapped with WMMA on the other buffer (TENSORcnt + barrier sync).
template <bool TDM>
__global__ void __launch_bounds__(256)
k_gemm_bt(const __bf16* __restrict__ A, const __bf16* __restrict__ B,
          const float* __restrict__ bias, int M, int N, int K, int relu,
          float* __restrict__ outF, __bf16* __restrict__ outB,
          float* __restrict__ scatF, __bf16* __restrict__ scatB,
          int seq, int hd, int batch_inner) {
  __shared__ __bf16 As[TDM ? 2 : 1][128][32];
  __shared__ __bf16 Bs[TDM ? 2 : 1][64][32];
  const int t = threadIdx.x;
  const int lane = t & 31, wid = t >> 5;
  const int l16 = lane & 15, hi = lane >> 4;
  const int waveM = wid >> 1, waveN = wid & 1;
  const int blockM = blockIdx.y * 128, blockN = blockIdx.x * 64;
  const int aRow = t >> 1, aK = (t & 1) * 16;   // 128 rows x 32 k
  const int bRow = t >> 2, bK = (t & 3) * 8;    // 64 rows x 32 k
  const int nsteps = K / 32;

  v8f acc[2][2];
  for (int mt = 0; mt < 2; ++mt)
    for (int nt = 0; nt < 2; ++nt) acc[mt][nt] = vzero();

  if constexpr (TDM) {
    if (wid == 0) {
      tdm_load_2d_bf16((unsigned)(size_t)&As[0][0][0], A + (size_t)blockM * K,
                       (unsigned)K, (unsigned)(M - blockM), 32, 128, (unsigned)K);
      tdm_load_2d_bf16((unsigned)(size_t)&Bs[0][0][0], B + (size_t)blockN * K,
                       (unsigned)K, (unsigned)(N - blockN), 32, 64, (unsigned)K);
    }
  }

  for (int s = 0; s < nsteps; ++s) {
    const int k0 = s * 32;
    int cur = 0;
    if constexpr (TDM) {
      cur = s & 1;
      if (wid == 0) __builtin_amdgcn_s_wait_tensorcnt(0);
      __syncthreads();                       // tile `cur` visible to all waves
      if (wid == 0 && s + 1 < nsteps) {      // DMA next tile while we compute
        int k1 = k0 + 32, nb = (s + 1) & 1;
        tdm_load_2d_bf16((unsigned)(size_t)&As[nb][0][0], A + (size_t)blockM * K + k1,
                         (unsigned)(K - k1), (unsigned)(M - blockM), 32, 128, (unsigned)K);
        tdm_load_2d_bf16((unsigned)(size_t)&Bs[nb][0][0], B + (size_t)blockN * K + k1,
                         (unsigned)(K - k1), (unsigned)(N - blockN), 32, 64, (unsigned)K);
      }
    } else {
      U4 z4 = {0, 0, 0, 0};
      U4 a0 = z4, a1 = z4;
      int gr = blockM + aRow;
      if (gr < M) {
        const U4* p = (const U4*)(A + (size_t)gr * K + k0 + aK);
        a0 = p[0]; a1 = p[1];
      }
      *(U4*)(&As[0][aRow][aK])     = a0;
      *(U4*)(&As[0][aRow][aK + 8]) = a1;
      *(U4*)(&Bs[0][bRow][bK]) = *(const U4*)(B + (size_t)(blockN + bRow) * K + k0 + bK);
      __syncthreads();
      if (k0 + 32 < K) {  // near-scope prefetch of next k-tile
        if (gr < M) __builtin_prefetch(A + (size_t)gr * K + k0 + 32 + aK, 0, 3);
        __builtin_prefetch(B + (size_t)(blockN + bRow) * K + k0 + 32 + bK, 0, 3);
      }
    }

    FragU af[2], bf[2];
#pragma unroll
    for (int mt = 0; mt < 2; ++mt) {
      int r = waveM * 32 + mt * 16 + l16;
      af[mt].q[0] = *(const U4*)(&As[cur][r][hi * 8]);
      af[mt].q[1] = *(const U4*)(&As[cur][r][16 + hi * 8]);
    }
#pragma unroll
    for (int nt = 0; nt < 2; ++nt) {
      int c = waveN * 32 + nt * 16 + l16;
      bf[nt].q[0] = *(const U4*)(&Bs[cur][c][hi * 8]);
      bf[nt].q[1] = *(const U4*)(&Bs[cur][c][16 + hi * 8]);
    }
#pragma unroll
    for (int mt = 0; mt < 2; ++mt)
#pragma unroll
      for (int nt = 0; nt < 2; ++nt)
        acc[mt][nt] = __builtin_amdgcn_wmma_f32_16x16x32_bf16(
            false, af[mt].v, false, bf[nt].v, (short)0, acc[mt][nt], false, false);
    __syncthreads();
  }

#pragma unroll
  for (int mt = 0; mt < 2; ++mt)
#pragma unroll
    for (int nt = 0; nt < 2; ++nt) {
      int col = blockN + waveN * 32 + nt * 16 + l16;
      float bv = bias ? bias[col] : 0.f;
#pragma unroll
      for (int r = 0; r < 8; ++r) {
        int row = blockM + waveM * 32 + mt * 16 + hi * 8 + r;
        if (row >= M) continue;
        float v = acc[mt][nt][r] + bv;
        if (relu) v = fmaxf(v, 0.f);
        size_t o = (size_t)row * N + col;
        if (outF) outF[o] = v;
        if (outB) outB[o] = (__bf16)v;
        if (scatF || scatB) {
          int bb, ii;
          if (batch_inner) { bb = row % BS; ii = row / BS; }
          else             { bb = row / seq; ii = row % seq; }
          int h = col / hd, dd = col % hd;
          size_t so = (((size_t)bb * NH + h) * seq + ii) * hd + dd;
          if (scatF) scatF[so] = v;
          if (scatB) scatB[so] = (__bf16)v;
        }
      }
    }
}

// ------------------------------------------------- fused attention (LDS scores)
// One block = (16 queries) x (b,h). Scores/probs live in LDS (320KB/WGP on CDNA5).
template <int HD>
__global__ void __launch_bounds__(256)
k_attn_fused(const __bf16* __restrict__ Qh, const __bf16* __restrict__ Kh,
             const __bf16* __restrict__ VhT, __bf16* __restrict__ Out,
             float scale, int dm) {
  constexpr int NF = HD / 32;   // k-fragments for scores
  constexpr int NT = HD / 16;   // n-tiles for P*V
  constexpr int KS = 8 / NT;    // k-split waves per n-tile
  __shared__ __bf16 sP[16][NQP];
  __shared__ float sPart[8][16][16];

  const int t = threadIdx.x, lane = t & 31, wid = t >> 5;
  const int l16 = lane & 15, hi = lane >> 4;
  const int qbase = blockIdx.x * 16;
  const int bh = blockIdx.y;
  const int b = bh >> 3, h = bh & 7;
  const size_t qkBase = (size_t)bh * NQ * HD;
  const U4 z4 = {0, 0, 0, 0};

  FragU aQ[NF];
  {
    int row = qbase + l16;
#pragma unroll
    for (int f = 0; f < NF; ++f) {
      if (row < NQ) {
        const __bf16* p = Qh + qkBase + (size_t)row * HD + f * 32;
        aQ[f].q[0] = *(const U4*)(p + hi * 8);
        aQ[f].q[1] = *(const U4*)(p + 16 + hi * 8);
      } else { aQ[f].q[0] = z4; aQ[f].q[1] = z4; }
    }
  }
  // phase 1: S = scale * Q K^T  -> LDS (bf16)
  for (int kt = wid; kt < NQP / 16; kt += 8) {
    int key = kt * 16 + l16;
    FragU bK[NF];
#pragma unroll
    for (int f = 0; f < NF; ++f) {
      if (key < NQ) {
        const __bf16* p = Kh + qkBase + (size_t)key * HD + f * 32;
        bK[f].q[0] = *(const U4*)(p + hi * 8);
        bK[f].q[1] = *(const U4*)(p + 16 + hi * 8);
      } else { bK[f].q[0] = z4; bK[f].q[1] = z4; }
    }
    v8f c = vzero();
#pragma unroll
    for (int f = 0; f < NF; ++f)
      c = __builtin_amdgcn_wmma_f32_16x16x32_bf16(false, aQ[f].v, false, bK[f].v,
                                                  (short)0, c, false, false);
#pragma unroll
    for (int r = 0; r < 8; ++r)
      sP[hi * 8 + r][kt * 16 + l16] = (__bf16)(c[r] * scale);
  }
  __syncthreads();
  // phase 2: rowwise softmax in LDS, write bf16 probs in place
  for (int rr = wid * 2; rr < wid * 2 + 2; ++rr) {
    float mx = -1e30f;
    for (int j = lane; j < NQ; j += 32) mx = fmaxf(mx, (float)sP[rr][j]);
    mx = wmax(mx);
    float sum = 0.f;
    for (int j = lane; j < NQ; j += 32) sum += __expf((float)sP[rr][j] - mx);
    sum = wsum(sum);
    float inv = 1.f / sum;
    for (int j = lane; j < NQ; j += 32)
      sP[rr][j] = (__bf16)(__expf((float)sP[rr][j] - mx) * inv);
    for (int j = NQ + lane; j < NQP; j += 32) sP[rr][j] = (__bf16)0.f;
  }
  __syncthreads();
  // phase 3: O = P * V (VhT is [bh, HD, NQP], zero padded)
  const int ntile = wid % NT, ks = wid / NT;
  v8f c = vzero();
  for (int kt = ks; kt < NQP / 32; kt += KS) {
    FragU a, bV;
    a.q[0] = *(const U4*)(&sP[l16][kt * 32 + hi * 8]);
    a.q[1] = *(const U4*)(&sP[l16][kt * 32 + 16 + hi * 8]);
    const __bf16* p = VhT + ((size_t)bh * HD + ntile * 16 + l16) * NQP + kt * 32;
    bV.q[0] = *(const U4*)(p + hi * 8);
    bV.q[1] = *(const U4*)(p + 16 + hi * 8);
    c = __builtin_amdgcn_wmma_f32_16x16x32_bf16(false, a.v, false, bV.v,
                                                (short)0, c, false, false);
  }
#pragma unroll
  for (int r = 0; r < 8; ++r) sPart[wid][hi * 8 + r][l16] = c[r];
  __syncthreads();
  if (wid < NT) {
#pragma unroll
    for (int r = 0; r < 8; ++r) {
      float v = 0.f;
#pragma unroll
      for (int s = 0; s < KS; ++s) v += sPart[wid + s * NT][hi * 8 + r][l16];
      int row = qbase + hi * 8 + r;
      if (row < NQ)
        Out[((size_t)row * BS + b) * dm + h * HD + wid * 16 + l16] = (__bf16)v;
    }
  }
}

// ------------------------------------------------- V transpose with key padding
__global__ void k_transpose_vh(const __bf16* __restrict__ in, __bf16* __restrict__ out, int hd) {
  size_t i = (size_t)blockIdx.x * blockDim.x + threadIdx.x;
  size_t total = (size_t)BS * NH * hd * NQP;
  if (i >= total) return;
  int k = (int)(i % NQP); size_t rest = i / NQP;
  int d = (int)(rest % hd); int bh = (int)(rest / hd);
  out[i] = (k < NQ) ? in[((size_t)bh * NQ + k) * hd + d] : (__bf16)0.f;
}

// ------------------------------------------------- top-2 partner select
__global__ void __launch_bounds__(256)
k_pair_select(const float* __restrict__ gious, const float* __restrict__ cxcys,
              int* __restrict__ sel0, int* __restrict__ second) {
  int g = blockIdx.x * 8 + (threadIdx.x >> 5);
  int lane = threadIdx.x & 31;
  int b = g / NQ, n = g % NQ;
  const float* row = gious + ((size_t)b * NQ + n) * NQ;
  float bv = -1e30f; int bi = 0;
  for (int j = lane; j < NQ; j += 32) {
    if (j == n) continue;
    float v = row[j];
    if (v > bv || (v == bv && j < bi)) { bv = v; bi = j; }
  }
#pragma unroll
  for (int o = 16; o >= 1; o >>= 1) {
    float ov = __shfl_xor(bv, o, 32);
    int oi = __shfl_xor(bi, o, 32);
    if (ov > bv || (ov == bv && oi < bi)) { bv = ov; bi = oi; }
  }
  if (lane == 0) {
    int m = n * BS + b;
    second[m] = bi;
    int idx1 = (int)cxcys[((size_t)b * NQ + n) * NQ + bi];
    sel0[m] = (idx1 == 0) ? 1 : 0;
  }
}

__global__ void k_build_pairs(const __bf16* __restrict__ q, const __bf16* __restrict__ v,
                              const int* __restrict__ sel0, const int* __restrict__ second,
                              __bf16* __restrict__ twoq, __bf16* __restrict__ twok,
                              __bf16* __restrict__ twov) {
  int m = blockIdx.x, d = threadIdx.x;
  int b = m % BS;
  int pm = second[m] * BS + b;
  int s = sel0[m];
  __bf16 qs = q[(size_t)m * D + d], q2 = q[(size_t)pm * D + d];
  __bf16 vs = v[(size_t)m * D + d], v2 = v[(size_t)pm * D + d];
  size_t o = (size_t)m * 2 * D;
  twoq[o + d] = s ? qs : q2;  twoq[o + D + d] = s ? q2 : qs;
  twok[o + d] = s ? qs : v2;  twok[o + D + d] = s ? v2 : qs;
  twov[o + d] = s ? vs : v2;  twov[o + D + d] = s ? v2 : vs;
}

// ------------------------------------------------- combine + double LayerNorm
__global__ void __launch_bounds__(256)
k_combine_ln(const float* __restrict__ tgt, const float* __restrict__ qpos,
             const float* __restrict__ tgt2, const float* __restrict__ tgt3full,
             const int* __restrict__ sel0,
             const float* __restrict__ g1, const float* __restrict__ b1,
             const float* __restrict__ g12, const float* __restrict__ b12,
             float* __restrict__ x_f, __bf16* __restrict__ xq_bf) {
  int m = blockIdx.x * 8 + (threadIdx.x >> 5);   // (n*BS + b)
  int lane = threadIdx.x & 31;
  int n = m / BS, b = m % BS;
  size_t tBase = ((size_t)b * NQ + n) * D;
  int off = sel0[m] ? 0 : D;
  float a1[8], a2[8], s1 = 0.f, s2 = 0.f;
#pragma unroll
  for (int i = 0; i < 8; ++i) {
    int d = lane * 8 + i;
    float tv = tgt[tBase + d];
    a1[i] = tv + tgt2[(size_t)m * D + d];
    a2[i] = tv + tgt3full[(size_t)m * 2 * D + off + d];
    s1 += a1[i]; s2 += a2[i];
  }
  float mu1 = wsum(s1) / D, mu2 = wsum(s2) / D;
  float v1 = 0.f, v2 = 0.f;
#pragma unroll
  for (int i = 0; i < 8; ++i) {
    float d1 = a1[i] - mu1, d2 = a2[i] - mu2;
    v1 += d1 * d1; v2 += d2 * d2;
  }
  float rs1 = rsqrtf(wsum(v1) / D + 1e-5f);
  float rs2 = rsqrtf(wsum(v2) / D + 1e-5f);
#pragma unroll
  for (int i = 0; i < 8; ++i) {
    int d = lane * 8 + i;
    float o1 = (a1[i] - mu1) * rs1 * g1[d] + b1[d];
    float o2 = (a2[i] - mu2) * rs2 * g12[d] + b12[d];
    float x = o1 + o2;
    x_f[tBase + d] = x;
    xq_bf[tBase + d] = (__bf16)(x + qpos[tBase + d]);
  }
}

// ------------------------------------------------- generic residual LayerNorm
__global__ void __launch_bounds__(256)
k_add_ln(const float* __restrict__ a, const float* __restrict__ r,
         const float* __restrict__ g, const float* __restrict__ beta,
         int M, float* __restrict__ outF, __bf16* __restrict__ outB) {
  int m = blockIdx.x * 8 + (threadIdx.x >> 5);
  if (m >= M) return;
  int lane = threadIdx.x & 31;
  float v[8], s = 0.f;
#pragma unroll
  for (int i = 0; i < 8; ++i) {
    int d = lane * 8 + i;
    v[i] = a[(size_t)m * D + d] + r[(size_t)m * D + d];
    s += v[i];
  }
  float mu = wsum(s) / D;
  float var = 0.f;
#pragma unroll
  for (int i = 0; i < 8; ++i) { float dv = v[i] - mu; var += dv * dv; }
  float rs = rsqrtf(wsum(var) / D + 1e-5f);
#pragma unroll
  for (int i = 0; i < 8; ++i) {
    int d = lane * 8 + i;
    float o = (v[i] - mu) * rs * g[d] + beta[d];
    if (outF) outF[(size_t)m * D + d] = o;
    if (outB) outB[(size_t)m * D + d] = (__bf16)o;
  }
}

// ------------------------------------------------- sampling loc + softmax(aw)
__global__ void __launch_bounds__(256)
k_loc_aw(const float* __restrict__ off_f, const float* __restrict__ aw_f,
         const float* __restrict__ ref, float* __restrict__ out_loc,
         float* __restrict__ out_aw) {
  int w = blockIdx.x * 8 + (threadIdx.x >> 5);  // (b,n,hpair)
  int lane = threadIdx.x & 31;
  int lp = lane & 15, half = lane >> 4;
  int h = (w % 4) * 2 + half;
  int bn = w / 4; int n = bn % NQ; int b = bn / NQ;
  int lvl = lp >> 2;
  size_t m = (size_t)b * NQ + n;
  float av = aw_f[m * 128 + h * 16 + lp];
  float mx = av;
#pragma unroll
  for (int o = 8; o >= 1; o >>= 1) mx = fmaxf(mx, __shfl_xor(mx, o, 16));
  float e = __expf(av - mx), sum = e;
#pragma unroll
  for (int o = 8; o >= 1; o >>= 1) sum += __shfl_xor(sum, o, 16);
  out_aw[m * 128 + h * 16 + lp] = e / sum;
  const float norm[4] = {100.f, 50.f, 25.f, 13.f};  // (W,H) per level, square
  float rx = ref[(m * NL + lvl) * 2 + 0];
  float ry = ref[(m * NL + lvl) * 2 + 1];
  size_t lo = (m * 128 + h * 16 + lp) * 2;
  out_loc[lo + 0] = rx + off_f[m * 256 + (h * 16 + lp) * 2 + 0] / norm[lvl];
  out_loc[lo + 1] = ry + off_f[m * 256 + (h * 16 + lp) * 2 + 1] / norm[lvl];
}

// ------------------------------------------------- deformable bilinear sampling
__global__ void __launch_bounds__(256)
k_deform_sample(const float* __restrict__ value, const float* __restrict__ loc,
                const float* __restrict__ aw, __bf16* __restrict__ outB) {
  int g = blockIdx.x * 8 + (threadIdx.x >> 5);  // (b*NH+h)*NQ + n
  int d = threadIdx.x & 31;                     // lane == head channel
  int n = g % NQ; int bh = g / NQ; int h = bh & 7; int b = bh >> 3;
  const int Hs[4] = {100, 50, 25, 13};
  const int St[4] = {0, 10000, 12500, 13125};
  size_t m = (size_t)b * NQ + n;
  const float* locp = loc + (m * NH + h) * 32;
  const float* awp  = aw + (m * NH + h) * 16;
  const float* vb = value + (size_t)bh * LS * 32;
  float acc = 0.f;
#pragma unroll
  for (int lvl = 0; lvl < 4; ++lvl) {
    int H = Hs[lvl], W = Hs[lvl];
    const float* vl = vb + (size_t)St[lvl] * 32;
#pragma unroll
    for (int p = 0; p < 4; ++p) {
      float x = locp[(lvl * 4 + p) * 2 + 0] * W - 0.5f;
      float y = locp[(lvl * 4 + p) * 2 + 1] * H - 0.5f;
      float x0f = floorf(x), y0f = floorf(y);
      float wx1 = x - x0f, wy1 = y - y0f;
      int x0 = (int)x0f, y0 = (int)y0f;
      float wt = awp[lvl * 4 + p], s = 0.f;
#pragma unroll
      for (int dy = 0; dy < 2; ++dy)
#pragma unroll
        for (int dx = 0; dx < 2; ++dx) {
          int xi = x0 + dx, yi = y0 + dy;
          bool valid = (xi >= 0) && (xi < W) && (yi >= 0) && (yi < H);
          int idx = min(max(yi, 0), H - 1) * W + min(max(xi, 0), W - 1);
          float v = valid ? vl[(size_t)idx * 32 + d] : 0.f;
          float cw = (dx ? wx1 : 1.f - wx1) * (dy ? wy1 : 1.f - wy1);
          s += cw * v;
        }
      acc += wt * s;
    }
  }
  outB[m * 256 + h * 32 + d] = (__bf16)acc;
}

// ================================================================ host driver
extern "C" void kernel_launch(void* const* d_in, const int* in_sizes, int n_in,
                              void* d_out, int out_size, void* d_ws, size_t ws_size,
                              hipStream_t stream) {
  (void)in_sizes; (void)n_in; (void)out_size; (void)ws_size;
  const float* tgt   = (const float*)d_in[0];
  const float* qpos  = (const float*)d_in[1];
  const float* ref   = (const float*)d_in[2];
  const float* src   = (const float*)d_in[3];
  const float* cxcys = (const float*)d_in[6];
  const float* gious = (const float*)d_in[7];
  const float* sa_wq = (const float*)d_in[8];  const float* sa_bq = (const float*)d_in[9];
  const float* sa_wk = (const float*)d_in[10]; const float* sa_bk = (const float*)d_in[11];
  const float* sa_wv = (const float*)d_in[12]; const float* sa_bv = (const float*)d_in[13];
  const float* sa_wo = (const float*)d_in[14]; const float* sa_bo = (const float*)d_in[15];
  const float* s2_wq = (const float*)d_in[16]; const float* s2_bq = (const float*)d_in[17];
  const float* s2_wk = (const float*)d_in[18]; const float* s2_bk = (const float*)d_in[19];
  const float* s2_wv = (const float*)d_in[20]; const float* s2_bv = (const float*)d_in[21];
  const float* s2_wo = (const float*)d_in[22]; const float* s2_bo = (const float*)d_in[23];
  const float* da_vw = (const float*)d_in[24]; const float* da_vb = (const float*)d_in[25];
  const float* da_ow = (const float*)d_in[26]; const float* da_ob = (const float*)d_in[27];
  const float* da_aw = (const float*)d_in[28]; const float* da_ab = (const float*)d_in[29];
  const float* da_outw = (const float*)d_in[30]; const float* da_outb = (const float*)d_in[31];
  const float* ln1_g = (const float*)d_in[32]; const float* ln1_b = (const float*)d_in[33];
  const float* ln12_g = (const float*)d_in[34]; const float* ln12_b = (const float*)d_in[35];
  const float* ln3_g = (const float*)d_in[36]; const float* ln3_b = (const float*)d_in[37];
  const float* ffn_w1 = (const float*)d_in[38]; const float* ffn_b1 = (const float*)d_in[39];
  const float* ffn_w2 = (const float*)d_in[40]; const float* ffn_b2 = (const float*)d_in[41];

  float* outX   = (float*)d_out;
  float* outLoc = outX + OUT_LOC;
  float* outAw  = outX + OUT_AW;

  char* ws = (char*)d_ws;
  size_t cur = 0;
  auto alloc = [&](size_t bytes) -> char* {
    char* p = ws + cur; cur = (cur + bytes + 255) & ~(size_t)255; return p;
  };
  const size_t M1 = (size_t)NQ * BS;  // 7200

  __bf16* q_bf    = (__bf16*)alloc(M1 * D * 2);
  __bf16* tgtT_bf = (__bf16*)alloc(M1 * D * 2);
  __bf16* wq_bf = (__bf16*)alloc(65536 * 2);  __bf16* wk_bf = (__bf16*)alloc(65536 * 2);
  __bf16* wv_bf = (__bf16*)alloc(65536 * 2);  __bf16* wo_bf = (__bf16*)alloc(65536 * 2);
  __bf16* w2q_bf = (__bf16*)alloc(262144 * 2); __bf16* w2k_bf = (__bf16*)alloc(262144 * 2);
  __bf16* w2v_bf = (__bf16*)alloc(262144 * 2); __bf16* w2o_bf = (__bf16*)alloc(262144 * 2);
  __bf16* davw_bf = (__bf16*)alloc(65536 * 2); __bf16* daow_bf = (__bf16*)alloc(65536 * 2);
  __bf16* daaw_bf = (__bf16*)alloc(32768 * 2); __bf16* daouw_bf = (__bf16*)alloc(65536 * 2);
  __bf16* fw1_bf = (__bf16*)alloc(262144 * 2); __bf16* fw2_bf = (__bf16*)alloc(262144 * 2);
  __bf16* qh_bf = (__bf16*)alloc((size_t)64 * NQ * 32 * 2);
  __bf16* kh_bf = (__bf16*)alloc((size_t)64 * NQ * 32 * 2);
  __bf16* vh_bf = (__bf16*)alloc((size_t)64 * NQ * 32 * 2);
  __bf16* vhT_bf = (__bf16*)alloc((size_t)64 * 32 * NQP * 2);
  __bf16* at1_bf = (__bf16*)alloc(M1 * D * 2);
  float*  tgt2_f = (float*)alloc(M1 * D * 4);
  __bf16* tgt2_bf = (__bf16*)alloc(M1 * D * 2);
  int* sel0 = (int*)alloc(M1 * 4);
  int* second = (int*)alloc(M1 * 4);
  __bf16* twoq = (__bf16*)alloc(M1 * 512 * 2);
  __bf16* twok = (__bf16*)alloc(M1 * 512 * 2);
  __bf16* twov = (__bf16*)alloc(M1 * 512 * 2);
  __bf16* q2h = (__bf16*)alloc((size_t)64 * NQ * 64 * 2);
  __bf16* k2h = (__bf16*)alloc((size_t)64 * NQ * 64 * 2);
  __bf16* v2h = (__bf16*)alloc((size_t)64 * NQ * 64 * 2);
  __bf16* v2hT = (__bf16*)alloc((size_t)64 * 64 * NQP * 2);
  __bf16* at2_bf = (__bf16*)alloc(M1 * 512 * 2);
  float* tgt3full = (float*)alloc(M1 * 512 * 4);
  float* x_f = (float*)alloc(M1 * D * 4);
  __bf16* xq_bf = (__bf16*)alloc(M1 * D * 2);
  __bf16* src_bf = (__bf16*)alloc((size_t)BS * LS * D * 2);
  float* value_f = (float*)alloc((size_t)BS * NH * LS * 32 * 4);
  float* off_f = (float*)alloc(M1 * 256 * 4);
  float* aw_f = (float*)alloc(M1 * 128 * 4);
  __bf16* samp_bf = (__bf16*)alloc(M1 * D * 2);
  float* ca_f = (float*)alloc(M1 * D * 4);
  float* x2_f = (float*)alloc(M1 * D * 4);
  __bf16* x2_bf = (__bf16*)alloc(M1 * D * 2);
  __bf16* ffnh_bf = (__bf16*)alloc(M1 * DFF * 2);
  float* ffn_f = (float*)alloc(M1 * D * 4);

  auto conv = [&](const float* s, __bf16* dst, size_t n) {
    k_f32_to_bf16<<<(unsigned)((n + 255) / 256), 256, 0, stream>>>(s, dst, n);
  };
  auto gemm = [&](bool tdm, const __bf16* A, const __bf16* B, const float* bias,
                  int M, int N, int K, int relu, float* oF, __bf16* oB,
                  float* sF, __bf16* sB, int seq, int hd, int binner) {
    dim3 grid(N / 64, (M + 127) / 128);
    if (tdm)
      k_gemm_bt<true><<<grid, 256, 0, stream>>>(A, B, bias, M, N, K, relu, oF, oB, sF, sB,
                                                seq, hd, binner);
    else
      k_gemm_bt<false><<<grid, 256, 0, stream>>>(A, B, bias, M, N, K, relu, oF, oB, sF, sB,
                                                 seq, hd, binner);
  };

  // weights -> bf16
  conv(sa_wq, wq_bf, 65536); conv(sa_wk, wk_bf, 65536);
  conv(sa_wv, wv_bf, 65536); conv(sa_wo, wo_bf, 65536);
  conv(s2_wq, w2q_bf, 262144); conv(s2_wk, w2k_bf, 262144);
  conv(s2_wv, w2v_bf, 262144); conv(s2_wo, w2o_bf, 262144);
  conv(da_vw, davw_bf, 65536); conv(da_ow, daow_bf, 65536);
  conv(da_aw, daaw_bf, 32768); conv(da_outw, daouw_bf, 65536);
  conv(ffn_w1, fw1_bf, 262144); conv(ffn_w2, fw2_bf, 262144);
  conv(src, src_bf, (size_t)BS * LS * D);

  // q = tgt+qpos, tgtT (both [NQ,BS,D] bf16)
  k_prep_q<<<(unsigned)(M1 * D / 256), 256, 0, stream>>>(tgt, qpos, q_bf, tgtT_bf);

  // self-attention #1
  gemm(false, q_bf, wq_bf, sa_bq, (int)M1, 256, 256, 0, nullptr, nullptr, nullptr, qh_bf, NQ, 32, 1);
  gemm(false, q_bf, wk_bf, sa_bk, (int)M1, 256, 256, 0, nullptr, nullptr, nullptr, kh_bf, NQ, 32, 1);
  gemm(false, tgtT_bf, wv_bf, sa_bv, (int)M1, 256, 256, 0, nullptr, nullptr, nullptr, vh_bf, NQ, 32, 1);
  {
    size_t n = (size_t)64 * 32 * NQP;
    k_transpose_vh<<<(unsigned)((n + 255) / 256), 256, 0, stream>>>(vh_bf, vhT_bf, 32);
  }
  k_attn_fused<32><<<dim3(57, 64), 256, 0, stream>>>(qh_bf, kh_bf, vhT_bf, at1_bf,
                                                     0.17677669529663687f /*1/sqrt(32)*/, 256);
  gemm(false, at1_bf, wo_bf, sa_bo, (int)M1, 256, 256, 0, tgt2_f, tgt2_bf, nullptr, nullptr, 0, 1, 0);

  // pairing + self-attention #2 (dm=512) -- TDM-staged GEMMs
  k_pair_select<<<900, 256, 0, stream>>>(gious, cxcys, sel0, second);
  k_build_pairs<<<(unsigned)M1, 256, 0, stream>>>(q_bf, tgt2_bf, sel0, second, twoq, twok, twov);
  gemm(true, twoq, w2q_bf, s2_bq, (int)M1, 512, 512, 0, nullptr, nullptr, nullptr, q2h, NQ, 64, 1);
  gemm(true, twok, w2k_bf, s2_bk, (int)M1, 512, 512, 0, nullptr, nullptr, nullptr, k2h, NQ, 64, 1);
  gemm(true, twov, w2v_bf, s2_bv, (int)M1, 512, 512, 0, nullptr, nullptr, nullptr, v2h, NQ, 64, 1);
  {
    size_t n = (size_t)64 * 64 * NQP;
    k_transpose_vh<<<(unsigned)((n + 255) / 256), 256, 0, stream>>>(v2h, v2hT, 64);
  }
  k_attn_fused<64><<<dim3(57, 64), 256, 0, stream>>>(q2h, k2h, v2hT, at2_bf, 0.125f, 512);
  gemm(true, at2_bf, w2o_bf, s2_bo, (int)M1, 512, 512, 0, tgt3full, nullptr, nullptr, nullptr, 0, 1, 0);

  // combine: out1+out2 -> x ([BS,NQ,D]) and x+qpos (bf16)
  k_combine_ln<<<900, 256, 0, stream>>>(tgt, qpos, tgt2_f, tgt3full, sel0,
                                        ln1_g, ln1_b, ln12_g, ln12_b, x_f, xq_bf);

  // deformable attention (value projection is the big one: M = BS*LS, TDM-staged)
  gemm(true, src_bf, davw_bf, da_vb, BS * LS, 256, 256, 0, nullptr, nullptr, value_f, nullptr, LS, 32, 0);
  gemm(false, xq_bf, daow_bf, da_ob, (int)M1, 256, 256, 0, off_f, nullptr, nullptr, nullptr, 0, 1, 0);
  gemm(false, xq_bf, daaw_bf, da_ab, (int)M1, 128, 256, 0, aw_f, nullptr, nullptr, nullptr, 0, 1, 0);
  k_loc_aw<<<3600, 256, 0, stream>>>(off_f, aw_f, ref, outLoc, outAw);
  k_deform_sample<<<(unsigned)(M1 * NH / 8), 256, 0, stream>>>(value_f, outLoc, outAw, samp_bf);
  gemm(false, samp_bf, daouw_bf, da_outb, (int)M1, 256, 256, 0, ca_f, nullptr, nullptr, nullptr, 0, 1, 0);
  k_add_ln<<<900, 256, 0, stream>>>(x_f, ca_f, ln1_g, ln1_b, (int)M1, x2_f, x2_bf);

  // FFN + final LN -> d_out (TDM-staged GEMMs)
  gemm(true, x2_bf, fw1_bf, ffn_b1, (int)M1, 1024, 256, 1, nullptr, ffnh_bf, nullptr, nullptr, 0, 1, 0);
  gemm(true, ffnh_bf, fw2_bf, ffn_b2, (int)M1, 256, 1024, 0, ffn_f, nullptr, nullptr, nullptr, 0, 1, 0);
  k_add_ln<<<900, 256, 0, stream>>>(x2_f, ffn_f, ln3_g, ln3_b, (int)M1, outX, nullptr);
}
